// Lstm_4587025072537
// MI455X (gfx1250) — matrix operand order repaired and checked by reading, not compiled, not run on hardware
//
#include <hip/hip_runtime.h>
#include <hip/hip_bf16.h>

// ---------------------------------------------------------------------------
// CDNA5 (gfx1250) bf16 WMMA LSTM language model.
//   embed:  [32768,256] @ [256,512]                -> bf16 embs
//   step t: [128,1536/2048] @ [*,4096] fused gates -> h (bf16), c (f32)
//   out:    [32768,1024] @ [1024,256] + b          -> f32 logits
// Weights pre-packed (group-of-4 fragments, K-major, contiguous per wave);
// L2-resident across all 256 timesteps. Each wave: 2 row-tiles x 4 col-tiles
// = 8 accumulators. K-loop unrolled x2 with ping-pong fragment buffers (no
// register copies); __launch_bounds__(128,1) so the ~180 live VGPRs fit
// without spilling (low occupancy + deep per-wave pipelining by design).
// ---------------------------------------------------------------------------

typedef __attribute__((ext_vector_type(16))) __bf16 v16bf;
typedef __attribute__((ext_vector_type(8)))  __bf16 v8bf;
typedef __attribute__((ext_vector_type(8)))  float  v8f;

#define LANES 32

// ---- WMMA wrapper ----------------------------------------------------------
__device__ __forceinline__ v8f wmma_bf16(v16bf a, v16bf b, v8f c) {
  return __builtin_amdgcn_wmma_f32_16x16x32_bf16(
      /*neg_a=*/false, a, /*neg_b=*/false, b,
      /*c_mod=*/(short)0, c, /*reuse_a=*/false, /*reuse_b=*/false);
}

// ---- A fragment: row-major bf16 [M x lda], tile at (m0, k0) ----------------
// ISA 16-bit A 16x32 layout: lane lo(0-15) = row m0+lo;
//   hi=0: elems 0..7 -> K=k0+0..7,  elems 8..15 -> K=k0+16..23
//   hi=1: elems 0..7 -> K=k0+8..15, elems 8..15 -> K=k0+24..31
__device__ __forceinline__ v16bf load_a_frag(const __bf16* __restrict__ A,
                                             int lda, int m0, int k0) {
  const int lane = threadIdx.x & (LANES - 1);
  const int lo = lane & 15;
  const int hi = lane >> 4;
  const __bf16* p = A + (size_t)(m0 + lo) * lda + k0 + hi * 8;
  v8bf a0 = *(const v8bf*)(p);
  v8bf a1 = *(const v8bf*)(p + 16);
  return __builtin_shufflevector(a0, a1, 0, 1, 2, 3, 4, 5, 6, 7,
                                 8, 9, 10, 11, 12, 13, 14, 15);
}

// ---- B fragment member from a group-of-4 base ------------------------------
// Packed layout: frag = (((ng * Ktiles + kt) * 4 + g) * 32 + lane) * 16 + e
// -> for one wave, per-kt data is one contiguous 4KB block.
__device__ __forceinline__ v16bf load_b_member(const __bf16* __restrict__ base,
                                               int g) {
  const int lane = threadIdx.x & (LANES - 1);
  return *(const v16bf*)(base + (size_t)g * 512 + lane * 16);
}

// ---- fragment bundle for one K-tile stage ----------------------------------
struct Frags {
  v16bf a[2];
  v16bf b[4];
};

__device__ __forceinline__ void mma8(v8f acc[2][4], const Frags& f) {
#pragma unroll
  for (int mt = 0; mt < 2; ++mt)
#pragma unroll
    for (int g = 0; g < 4; ++g)
      acc[mt][g] = wmma_bf16(f.a[mt], f.b[g], acc[mt][g]);
}

// ---- f32 -> bf16 convert ---------------------------------------------------
__global__ void f32_to_bf16_kernel(const float* __restrict__ in,
                                   __bf16* __restrict__ out, size_t n) {
  size_t i = (size_t)blockIdx.x * blockDim.x + threadIdx.x;
  if (i < n) out[i] = (__bf16)in[i];
}

// ---- pack f32 weight [K x N] row-major into grouped B fragment layout ------
// mode 0 (generic): group ng covers n-tiles ng*4..ng*4+3 ; n=(ng*4+g)*16+lo
// mode 1 (lstm):    group ng = unit tile, member g = gate ; n=g*1024+ng*16+lo
__global__ void pack_b_kernel(const float* __restrict__ W,
                              __bf16* __restrict__ Bp, int K, int N,
                              int mode) {
  int idx = blockIdx.x * blockDim.x + threadIdx.x;
  int Ktiles = K >> 5;
  int total = Ktiles * (N >> 4) * LANES * 16;
  if (idx >= total) return;
  int e    = idx & 15;
  int lane = (idx >> 4) & 31;
  int g    = (idx >> 9) & 3;
  int tmp  = idx >> 11;
  int kt = tmp % Ktiles;
  int ng = tmp / Ktiles;
  int lo = lane & 15;
  int n = mode ? (g * 1024 + ng * 16 + lo) : ((ng * 4 + g) * 16 + lo);
  int kk = kt * 32 + ((e & 8) << 1) + ((lane >> 4) << 3) + (e & 7);  // lanes 0-15: K 0..7,16..23; 16-31: 8..15,24..31
  Bp[idx] = (__bf16)W[(size_t)kk * N + n];
}

// ---- generic WMMA GEMM: C[M,N] = A[M,K](bf16) @ Bp + bias ------------------
// grid.x = M/32 ; grid.y = N/256 ; block 128 = 4 waves; wave = 32 rows x 64
// cols. Ktiles must be even (true for all uses here: 8, 32).
template <bool OUT_F32>
__global__ __launch_bounds__(128, 1) void wmma_gemm_kernel(
    const __bf16* __restrict__ A, const __bf16* __restrict__ Bp,
    const float* __restrict__ bias, void* __restrict__ Cout, int N, int K) {
  const int wave = threadIdx.x >> 5;
  const int m0 = blockIdx.x * 32;
  const int ng = blockIdx.y * 4 + wave;  // group of 4 n-tiles
  const int Ktiles = K >> 5;
  const __bf16* bbase = Bp + (size_t)ng * Ktiles * 2048;

  v8f acc[2][4];
#pragma unroll
  for (int mt = 0; mt < 2; ++mt)
#pragma unroll
    for (int i = 0; i < 4; ++i) acc[mt][i] = (v8f){};

  auto load_stage = [&](Frags& f, int kt) {
    f.a[0] = load_a_frag(A, K, m0, kt * 32);
    f.a[1] = load_a_frag(A, K, m0 + 16, kt * 32);
    const __bf16* bp = bbase + (size_t)kt * 2048;
#pragma unroll
    for (int i = 0; i < 4; ++i) f.b[i] = load_b_member(bp, i);
  };

  Frags fA, fB;
  load_stage(fA, 0);
  for (int kt = 0; kt < Ktiles; kt += 2) {
    load_stage(fB, kt + 1);   // in flight during fA's WMMAs
    mma8(acc, fA);
    if (kt + 2 < Ktiles) load_stage(fA, kt + 2);  // in flight during fB's
    mma8(acc, fB);
  }

  const int lane = threadIdx.x & 31;
  const int lo = lane & 15, hi = lane >> 4;
#pragma unroll
  for (int mt = 0; mt < 2; ++mt)
#pragma unroll
    for (int i = 0; i < 4; ++i) {
      const int col = (ng * 4 + i) * 16 + lo;
      const float bv = bias ? bias[col] : 0.0f;
#pragma unroll
      for (int r = 0; r < 8; ++r) {
        const int row = m0 + mt * 16 + r + hi * 8;
        const float v = acc[mt][i][r] + bv;
        if (OUT_F32)
          ((float*)Cout)[(size_t)row * N + col] = v;
        else
          ((__bf16*)Cout)[(size_t)row * N + col] = (__bf16)v;
      }
    }
}

// ---- fused LSTM step: lin = [A1 ; A2] @ Wp + b ; gates ; c,h update --------
// 128 rows, hidden 1024 (4096 gemm cols). 256 waves: grid 64 x block 128.
// wave wid: mg = wid>>6 (rows mg*32..mg*32+31), u = wid&63 (unit tile).
// Ktiles even for both layers (48, 64).
__global__ __launch_bounds__(128, 1) void lstm_step_kernel(
    const __bf16* __restrict__ A1, int K1, const __bf16* __restrict__ A2,
    int K2, const __bf16* __restrict__ Wp, const float* __restrict__ bias,
    float* __restrict__ c, __bf16* __restrict__ h, __bf16* __restrict__ hs) {
  const int wid = blockIdx.x * (blockDim.x >> 5) + (threadIdx.x >> 5);
  const int mg = wid >> 6;
  const int u = wid & 63;
  const int m0 = mg * 32;
  const int K1t = K1 >> 5;
  const int Ktiles = (K1 + K2) >> 5;
  const __bf16* bbase = Wp + (size_t)u * Ktiles * 2048;

  v8f acc[2][4];
#pragma unroll
  for (int mt = 0; mt < 2; ++mt)
#pragma unroll
    for (int g = 0; g < 4; ++g) acc[mt][g] = (v8f){};

  auto load_stage = [&](Frags& f, int kt) {  // wave-uniform A1/A2 switch
    if (kt < K1t) {
      f.a[0] = load_a_frag(A1, K1, m0, kt * 32);
      f.a[1] = load_a_frag(A1, K1, m0 + 16, kt * 32);
    } else {
      f.a[0] = load_a_frag(A2, K2, m0, (kt - K1t) * 32);
      f.a[1] = load_a_frag(A2, K2, m0 + 16, (kt - K1t) * 32);
    }
    const __bf16* bp = bbase + (size_t)kt * 2048;
#pragma unroll
    for (int g = 0; g < 4; ++g) f.b[g] = load_b_member(bp, g);
  };

  Frags fA, fB;
  load_stage(fA, 0);
  for (int kt = 0; kt < Ktiles; kt += 2) {
    load_stage(fB, kt + 1);   // in flight during fA's WMMAs
    mma8(acc, fA);
    if (kt + 2 < Ktiles) load_stage(fA, kt + 2);  // in flight during fB's
    mma8(acc, fB);
  }

  const int lane = threadIdx.x & 31;
  const int lo = lane & 15, hi = lane >> 4;
  const int unit = u * 16 + lo;
  const float bF = bias[unit];
  const float bI = bias[1024 + unit];
  const float bO = bias[2048 + unit];
  const float bG = bias[3072 + unit];
#pragma unroll
  for (int mt = 0; mt < 2; ++mt) {
#pragma unroll
    for (int r = 0; r < 8; ++r) {
      const int row = m0 + mt * 16 + r + hi * 8;
      const size_t off = (size_t)row * 1024 + unit;
      const float f  = 1.0f / (1.0f + __expf(-(acc[mt][0][r] + bF)));
      const float ig = 1.0f / (1.0f + __expf(-(acc[mt][1][r] + bI)));
      const float o  = 1.0f / (1.0f + __expf(-(acc[mt][2][r] + bO)));
      const float g  = tanhf(acc[mt][3][r] + bG);
      const float cn = f * c[off] + ig * g;
      c[off] = cn;
      const float hv = o * tanhf(cn);
      h[off] = (__bf16)hv;
      if (hs) hs[off] = (__bf16)hv;
    }
  }
}

// ---------------------------------------------------------------------------
extern "C" void kernel_launch(void* const* d_in, const int* in_sizes, int n_in,
                              void* d_out, int out_size, void* d_ws,
                              size_t ws_size, hipStream_t stream) {
  (void)in_sizes; (void)n_in; (void)out_size; (void)ws_size;
  const float* inputs = (const float*)d_in[0];  // [256,128,256]
  const float* emb    = (const float*)d_in[1];  // [256,512]
  const float* w0     = (const float*)d_in[2];  // [1536,4096]
  const float* b0     = (const float*)d_in[3];  // [4096]
  const float* w1     = (const float*)d_in[4];  // [2048,4096]
  const float* b1     = (const float*)d_in[5];  // [4096]
  const float* out_w  = (const float*)d_in[6];  // [1024,256]
  const float* out_b  = (const float*)d_in[7];  // [256]
  float* out = (float*)d_out;                   // [32768,256]

  const int T = 256, B = 128, V = 256, E = 512, H = 1024;

  char* ws = (char*)d_ws;
  size_t off = 0;
  auto alloc = [&](size_t bytes) -> char* {
    char* p = ws + off;
    off += (bytes + 255) & ~(size_t)255;
    return p;
  };
  __bf16* inBf  = (__bf16*)alloc((size_t)T * B * V * 2);        // 16 MB
  __bf16* embBf = (__bf16*)alloc((size_t)T * B * E * 2);        // 32 MB
  __bf16* embP  = (__bf16*)alloc((size_t)V * E * 2);
  __bf16* w0P   = (__bf16*)alloc((size_t)(E + H) * 4 * H * 2);  // 12 MB
  __bf16* w1P   = (__bf16*)alloc((size_t)(2 * H) * 4 * H * 2);  // 16 MB
  __bf16* owP   = (__bf16*)alloc((size_t)H * V * 2);
  __bf16* h0p[2] = {(__bf16*)alloc((size_t)B * H * 2),
                    (__bf16*)alloc((size_t)B * H * 2)};
  __bf16* h1p[2] = {(__bf16*)alloc((size_t)B * H * 2),
                    (__bf16*)alloc((size_t)B * H * 2)};
  float* c0 = (float*)alloc((size_t)B * H * 4);
  float* c1 = (float*)alloc((size_t)B * H * 4);
  __bf16* hsB = (__bf16*)alloc((size_t)T * B * H * 2);          // 64 MB

  // zero initial states (graph-capturable)
  hipMemsetAsync(h0p[0], 0, (size_t)B * H * 2, stream);
  hipMemsetAsync(h1p[0], 0, (size_t)B * H * 2, stream);
  hipMemsetAsync(c0, 0, (size_t)B * H * 4, stream);
  hipMemsetAsync(c1, 0, (size_t)B * H * 4, stream);

  // convert activations, pack weights (weights stay L2-resident afterwards)
  {
    size_t n = (size_t)T * B * V;
    f32_to_bf16_kernel<<<(unsigned)((n + 255) / 256), 256, 0, stream>>>(
        inputs, inBf, n);
  }
  auto pack = [&](const float* W, __bf16* P, int K, int N, int mode) {
    int total = (K >> 5) * (N >> 4) * LANES * 16;
    pack_b_kernel<<<(total + 255) / 256, 256, 0, stream>>>(W, P, K, N, mode);
  };
  pack(emb, embP, V, E, 0);
  pack(w0, w0P, E + H, 4 * H, 1);
  pack(w1, w1P, 2 * H, 4 * H, 1);
  pack(out_w, owP, H, V, 0);

  // embedding GEMM: [32768,256] @ [256,512] -> bf16
  {
    dim3 grid(T * B / 32, E / 256);
    wmma_gemm_kernel<false><<<grid, 128, 0, stream>>>(inBf, embP, nullptr,
                                                      embBf, E, V);
  }

  // 256 sequential timesteps, 2 fused layer kernels each
  int cur = 0;
  for (int t = 0; t < T; ++t) {
    const __bf16* x = embBf + (size_t)t * B * E;
    lstm_step_kernel<<<64, 128, 0, stream>>>(x, E, h0p[cur], H, w0P, b0, c0,
                                             h0p[1 - cur], nullptr);
    lstm_step_kernel<<<64, 128, 0, stream>>>(h0p[1 - cur], H, h1p[cur], H,
                                             w1P, b1, c1, h1p[1 - cur],
                                             hsB + (size_t)t * B * H);
    cur ^= 1;
  }

  // output GEMM: [32768,1024] @ [1024,256] + out_b -> f32 logits
  {
    dim3 grid(T * B / 32, 1);
    wmma_gemm_kernel<true><<<grid, 128, 0, stream>>>(hsB, owP, out_b, out, V,
                                                     H);
  }
}
